// MultiHeadAttention_32822140076309
// MI455X (gfx1250) — compile-verified, matrix operand order checked by reading, not run on hardware
//
#include <hip/hip_runtime.h>
#include <hip/hip_bf16.h>

// MHA forward, MI455X (gfx1250, wave32, WMMA).
// B=2, S=2048, D=1024, H=16, hd=64.
//
// Roofline: ~103 GFLOP total; essential HBM traffic ~84 MB (~3.6us @23.3TB/s);
// the 48MB of f16 intermediates below fit in the 192MB L2, so the problem is
// matrix-pipe bound -> f16 WMMA with f32 accumulation, LDS-staged operands
// filled with gfx1250 async global->LDS copies.
//
//   ws[ 0..8M)   xh   : x as f16           [4096,1024]
//   ws[ 8..10M)  wqh  : Wq as f16          [1024,1024]
//   ws[10..12M)  wkh  : Wk as f16
//   ws[12..14M)  wvh  : Wv as f16
//   ws[14..16M)  woh  : Wo as f16
//   ws[16..24M)  qh   : Q*0.125 f16        [b,h,s,hd]
//   ws[24..32M)  kh   : K f16              [b,h,s,hd]
//   ws[32..40M)  vt   : V f16 TRANSPOSED   [b,h,hd,s]
//   ws[40..48M)  oh   : attn out f16       [4096,1024]

typedef __attribute__((ext_vector_type(16))) _Float16 v16h;
typedef __attribute__((ext_vector_type(8)))  _Float16 v8h;
typedef __attribute__((ext_vector_type(4)))  _Float16 v4h;
typedef __attribute__((ext_vector_type(8)))  float    v8f;

// exact parameter types of __builtin_amdgcn_global_load_async_to_lds_b128
typedef int v4i_ __attribute__((vector_size(16)));
typedef v4i_ __attribute__((address_space(1)))* gas1_i4p;
typedef v4i_ __attribute__((address_space(3)))* las3_i4p;

#define DMODEL 1024
#define SEQ    2048
#define NB     2
#define NH     16
#define HD     64
#define MTOT   (NB * SEQ)

// WMMA fragment convention (wave32):
//  A/B frag (16x32 f16): lane L holds row (L&15), 16 contiguous f16 at
//                        k-offset ((L>>4)*16).
//  C/D tile (16x16 f32): VGPR r, lane L -> M = r + 8*(L>>4), N = (L&15).

__device__ __forceinline__ v8f wmma16(v16h a, v16h b, v8f c) {
    return __builtin_amdgcn_wmma_f32_16x16x32_f16(false, a, false, b, (short)0, c, false, false);
}

// ---- gfx1250 async global->LDS copy (16B per lane), fallback: reg bounce ---
__device__ __forceinline__ void cpB16(_Float16* lds_dst, const _Float16* gsrc) {
#if defined(__has_builtin) && __has_builtin(__builtin_amdgcn_global_load_async_to_lds_b128)
    __builtin_amdgcn_global_load_async_to_lds_b128(
        (gas1_i4p)gsrc, (las3_i4p)lds_dst, /*offset=*/0, /*cpol=*/0);
#else
    *(v8h*)lds_dst = *(const v8h*)gsrc;
#endif
}

__device__ __forceinline__ void wait_async0() {
#if defined(__has_builtin) && __has_builtin(__builtin_amdgcn_s_wait_asynccnt)
    __builtin_amdgcn_s_wait_asynccnt(0);
#else
    asm volatile("s_wait_asynccnt 0" ::: "memory");
#endif
}

__device__ __forceinline__ void wait_ds0() {
#if defined(__has_builtin) && __has_builtin(__builtin_amdgcn_s_wait_dscnt)
    __builtin_amdgcn_s_wait_dscnt(0);
#else
    asm volatile("s_wait_dscnt 0" ::: "memory");
#endif
}

// ---------------------------------------------------------------- convert ---
__global__ __launch_bounds__(256) void k_cvt_f32_f16(const float* __restrict__ src,
                                                     _Float16* __restrict__ dst, int n4) {
    int i = blockIdx.x * blockDim.x + threadIdx.x;
    if (i < n4) {
        float4 f = ((const float4*)src)[i];
        v4h h = { (_Float16)f.x, (_Float16)f.y, (_Float16)f.z, (_Float16)f.w };
        ((v4h*)dst)[i] = h;
    }
}

// ------------------------------------------------------------ QKV project ---
// y = x @ W.T. Block = 8 waves sharing one 64-col W panel; the 64x32 W k-slice
// is staged in LDS (double-buffered, async-filled). Each wave: 16(M)x64(N).
// 3 matrices * 512 blocks each = 1536 blocks.
__global__ __launch_bounds__(256) void k_proj_qkv(const _Float16* __restrict__ xh,
                                                  const _Float16* __restrict__ wqh,
                                                  const _Float16* __restrict__ wkh,
                                                  const _Float16* __restrict__ wvh,
                                                  _Float16* __restrict__ qh,
                                                  _Float16* __restrict__ kh,
                                                  _Float16* __restrict__ vt) {
    __shared__ __align__(128) _Float16 Bl[2][64 * 32];   // 2 x 4KB W slices

    const int tid   = threadIdx.x;
    const int lane  = tid & 31;
    const int ln    = lane & 15;
    const int hi    = lane >> 4;
    const int khalf = hi * 16;
    const int w     = tid >> 5;

    const int which = blockIdx.x / 512;          // 0=Q 1=K 2=V (block-uniform)
    const int rem   = blockIdx.x % 512;
    const int mt    = (rem * 8 + w) & 255;       // per-wave m tile
    const int nt    = rem >> 5;                  // block-uniform n panel (0..15)
    const _Float16* W = (which == 0) ? wqh : (which == 1) ? wkh : wvh;

    // cooperative fill mapping: thread -> (row 0..63, 8-elem chunk 0..3)
    const int frow = tid >> 2;
    const int fcol = (tid & 3) * 8;
    const _Float16* wsrc = W + (size_t)(nt * 64 + frow) * DMODEL + fcol;
    _Float16* const ldst0 = &Bl[0][frow * 32 + fcol];
    _Float16* const ldst1 = &Bl[1][frow * 32 + fcol];

    const _Float16* arow = xh + (size_t)(mt * 16 + ln) * DMODEL + khalf;

    cpB16(ldst0, wsrc);                           // prefill k-step 0
    v16h a_cur = *(const v16h*)(arow);
    v8f acc[4] = {};

    for (int kt = 0, st = 0; kt < DMODEL; kt += 32, st ^= 1) {
        wait_async0();
        __syncthreads();                          // Bl[st] ready for everyone
        int ktn = (kt + 32 < DMODEL) ? kt + 32 : kt;
        cpB16(st ? ldst0 : ldst1, wsrc + ktn);    // async fill next slice
        __builtin_prefetch(arow + ktn + 32, 0, 3);
        v16h a_nxt = *(const v16h*)(arow + ktn);

        v16h b0 = *(const v16h*)&Bl[st][(0 * 16 + ln) * 32 + khalf];
        v16h b1 = *(const v16h*)&Bl[st][(1 * 16 + ln) * 32 + khalf];
        v16h b2 = *(const v16h*)&Bl[st][(2 * 16 + ln) * 32 + khalf];
        v16h b3 = *(const v16h*)&Bl[st][(3 * 16 + ln) * 32 + khalf];
        acc[0] = wmma16(a_cur, b0, acc[0]);
        acc[1] = wmma16(a_cur, b1, acc[1]);
        acc[2] = wmma16(a_cur, b2, acc[2]);
        acc[3] = wmma16(a_cur, b3, acc[3]);
        a_cur = a_nxt;
    }

    const int bb = (mt * 16) >> 11;               // batch (tile never straddles)
    const int s0 = (mt * 16 + 8 * hi) & 2047;
#pragma unroll
    for (int j = 0; j < 4; ++j) {
        int n = nt * 64 + j * 16 + ln;
        int h = n >> 6, d = n & 63;
        if (which == 2) {
            // Vt[bh][d][s]: lane's 8 values are consecutive in s -> one 16B store
            v8h pk;
#pragma unroll
            for (int r = 0; r < 8; ++r) pk[r] = (_Float16)acc[j][r];
            *(v8h*)(vt + ((size_t)(bb * NH + h) * HD + d) * SEQ + s0) = pk;
        } else {
#pragma unroll
            for (int r = 0; r < 8; ++r) {
                size_t bhs = (size_t)(bb * NH + h) * SEQ + (s0 + r);
                if (which == 0) qh[bhs * HD + d] = (_Float16)(acc[j][r] * 0.125f);
                else            kh[bhs * HD + d] = (_Float16)acc[j][r];
            }
        }
    }
}

// ------------------------------------------------------- flash attention ----
// Block = 4 waves = 64 query rows; kv blocks of 32 keys. K tile (32x64) and
// V tile (64x32, pre-transposed) staged in LDS (double-buffered, async),
// shared by all 4 waves. Trip count is block-uniform (causal mask handles the
// ragged edge) so EXEC is all-ones at every WMMA and barriers are legal.
__global__ __launch_bounds__(128) void k_attn(const _Float16* __restrict__ qh,
                                              const _Float16* __restrict__ kh,
                                              const _Float16* __restrict__ vt,
                                              _Float16* __restrict__ oh) {
    __shared__ __align__(128) _Float16 Kl[2][32 * 64];   // 2 x 4KB
    __shared__ __align__(128) _Float16 Vl[2][64 * 32];   // 2 x 4KB
    __shared__ __align__(64)  _Float16 plds[4 * 16 * 32];

    const int tid   = threadIdx.x;
    const int lane  = tid & 31;
    const int ln    = lane & 15;
    const int hi    = lane >> 4;
    const int khalf = hi * 16;
    const int w     = tid >> 5;

    const int qb = blockIdx.x & 31;
    const int bh = blockIdx.x >> 5;
    const int b  = bh >> 4, h = bh & 15;

    const _Float16* Q = qh + (size_t)bh * SEQ * HD;
    const _Float16* K = kh + (size_t)bh * SEQ * HD;
    const _Float16* V = vt + (size_t)bh * HD * SEQ;

    const int q0 = qb * 64 + w * 16;

    // cooperative fill mappings
    const int kr = tid >> 2, kc = (tid & 3) * 16;   // K: 32 rows x 64 -> 32B/thread
    const int vr = tid >> 1, vc = (tid & 1) * 16;   // V: 64 rows x 32 -> 32B/thread
    const _Float16* ksrc = K + (size_t)kr * HD + kc;
    const _Float16* vsrc = V + (size_t)vr * SEQ + vc;

    // Q fragments are kv-loop invariant
    v16h aq0 = *(const v16h*)(Q + (size_t)(q0 + ln) * HD + 0  + khalf);
    v16h aq1 = *(const v16h*)(Q + (size_t)(q0 + ln) * HD + 32 + khalf);

    v8f   oacc[4] = {};
    float m_i[8], l_i[8];
#pragma unroll
    for (int r = 0; r < 8; ++r) { m_i[r] = -1e30f; l_i[r] = 0.0f; }

    // prefill kv block 0
    cpB16(&Kl[0][kr * HD + kc],     ksrc);
    cpB16(&Kl[0][kr * HD + kc + 8], ksrc + 8);
    cpB16(&Vl[0][vr * 32 + vc],     vsrc);
    cpB16(&Vl[0][vr * 32 + vc + 8], vsrc + 8);

    const int kv_end = qb * 64 + 64;
    for (int kv0 = 0, st = 0; kv0 < kv_end; kv0 += 32, st ^= 1) {
        wait_async0();
        __syncthreads();                          // Kl/Vl[st] ready
        int kvn = (kv0 + 32 < kv_end) ? kv0 + 32 : kv0;
        int nb = st ^ 1;
        cpB16(&Kl[nb][kr * HD + kc],     ksrc + (size_t)kvn * HD);
        cpB16(&Kl[nb][kr * HD + kc + 8], ksrc + (size_t)kvn * HD + 8);
        cpB16(&Vl[nb][vr * 32 + vc],     vsrc + kvn);
        cpB16(&Vl[nb][vr * 32 + vc + 8], vsrc + kvn + 8);

        // ---- scores: 16 q x 32 keys over hd=64 ------------------------------
        v8f stl[2];
#pragma unroll
        for (int jn = 0; jn < 2; ++jn) {
            v8f s = {};
            v16h b0 = *(const v16h*)&Kl[st][(jn * 16 + ln) * HD + khalf];
            v16h b1 = *(const v16h*)&Kl[st][(jn * 16 + ln) * HD + 32 + khalf];
            s = wmma16(aq0, b0, s);
            s = wmma16(aq1, b1, s);
            stl[jn] = s;
        }

        // V fragments issued early: ds latency hides under softmax VALU work
        v16h bv0 = *(const v16h*)&Vl[st][(0 * 16 + ln) * 32 + khalf];
        v16h bv1 = *(const v16h*)&Vl[st][(1 * 16 + ln) * 32 + khalf];
        v16h bv2 = *(const v16h*)&Vl[st][(2 * 16 + ln) * 32 + khalf];
        v16h bv3 = *(const v16h*)&Vl[st][(3 * 16 + ln) * 32 + khalf];

        // ---- causal mask + online softmax -----------------------------------
#pragma unroll
        for (int r = 0; r < 8; ++r) {
            int row = q0 + r + 8 * hi;
            float s0 = stl[0][r]; if (kv0 + ln      > row) s0 = -1e30f;
            float s1 = stl[1][r]; if (kv0 + 16 + ln > row) s1 = -1e30f;

            float lm = fmaxf(s0, s1);
            lm = fmaxf(lm, __shfl_xor(lm, 1, 32));
            lm = fmaxf(lm, __shfl_xor(lm, 2, 32));
            lm = fmaxf(lm, __shfl_xor(lm, 4, 32));
            lm = fmaxf(lm, __shfl_xor(lm, 8, 32));

            float mnew  = fmaxf(m_i[r], lm);
            float scale = __expf(m_i[r] - mnew);
            float p0 = __expf(s0 - mnew);
            float p1 = __expf(s1 - mnew);
            float rs = p0 + p1;
            rs += __shfl_xor(rs, 1, 32);
            rs += __shfl_xor(rs, 2, 32);
            rs += __shfl_xor(rs, 4, 32);
            rs += __shfl_xor(rs, 8, 32);
            l_i[r] = l_i[r] * scale + rs;
            m_i[r] = mnew;
#pragma unroll
            for (int jd = 0; jd < 4; ++jd) oacc[jd][r] *= scale;

            plds[w * 512 + (r + 8 * hi) * 32 + ln]      = (_Float16)p0;
            plds[w * 512 + (r + 8 * hi) * 32 + 16 + ln] = (_Float16)p1;
        }
        wait_ds0();                               // P write -> read, same wave

        // ---- O += P(16x32) @ V(32x64) ---------------------------------------
        v16h ap = *(const v16h*)(plds + w * 512 + ln * 32 + khalf);
        oacc[0] = wmma16(ap, bv0, oacc[0]);
        oacc[1] = wmma16(ap, bv1, oacc[1]);
        oacc[2] = wmma16(ap, bv2, oacc[2]);
        oacc[3] = wmma16(ap, bv3, oacc[3]);
    }

#pragma unroll
    for (int r = 0; r < 8; ++r) {
        int row = q0 + r + 8 * hi;
        size_t t = (size_t)b * SEQ + row;
        float inv = 1.0f / l_i[r];
#pragma unroll
        for (int jd = 0; jd < 4; ++jd) {
            int d = jd * 16 + ln;
            oh[t * DMODEL + h * HD + d] = (_Float16)(oacc[jd][r] * inv);
        }
    }
}

// --------------------------------------------------------- output project ---
// out = oh @ Wo.T, fp32 result; same LDS-staged structure. 512 blocks.
__global__ __launch_bounds__(256) void k_proj_out(const _Float16* __restrict__ oh,
                                                  const _Float16* __restrict__ woh,
                                                  float* __restrict__ out) {
    __shared__ __align__(128) _Float16 Bl[2][64 * 32];

    const int tid   = threadIdx.x;
    const int lane  = tid & 31;
    const int ln    = lane & 15;
    const int hi    = lane >> 4;
    const int khalf = hi * 16;
    const int w     = tid >> 5;

    const int rem = blockIdx.x;
    const int mt  = (rem * 8 + w) & 255;
    const int nt  = rem >> 5;

    const int frow = tid >> 2;
    const int fcol = (tid & 3) * 8;
    const _Float16* wsrc = woh + (size_t)(nt * 64 + frow) * DMODEL + fcol;
    _Float16* const ldst0 = &Bl[0][frow * 32 + fcol];
    _Float16* const ldst1 = &Bl[1][frow * 32 + fcol];

    const _Float16* arow = oh + (size_t)(mt * 16 + ln) * DMODEL + khalf;

    cpB16(ldst0, wsrc);
    v16h a_cur = *(const v16h*)(arow);
    v8f acc[4] = {};

    for (int kt = 0, st = 0; kt < DMODEL; kt += 32, st ^= 1) {
        wait_async0();
        __syncthreads();
        int ktn = (kt + 32 < DMODEL) ? kt + 32 : kt;
        cpB16(st ? ldst0 : ldst1, wsrc + ktn);
        __builtin_prefetch(arow + ktn + 32, 0, 3);
        v16h a_nxt = *(const v16h*)(arow + ktn);

        v16h b0 = *(const v16h*)&Bl[st][(0 * 16 + ln) * 32 + khalf];
        v16h b1 = *(const v16h*)&Bl[st][(1 * 16 + ln) * 32 + khalf];
        v16h b2 = *(const v16h*)&Bl[st][(2 * 16 + ln) * 32 + khalf];
        v16h b3 = *(const v16h*)&Bl[st][(3 * 16 + ln) * 32 + khalf];
        acc[0] = wmma16(a_cur, b0, acc[0]);
        acc[1] = wmma16(a_cur, b1, acc[1]);
        acc[2] = wmma16(a_cur, b2, acc[2]);
        acc[3] = wmma16(a_cur, b3, acc[3]);
        a_cur = a_nxt;
    }

#pragma unroll
    for (int j = 0; j < 4; ++j) {
        int n = nt * 64 + j * 16 + ln;
#pragma unroll
        for (int r = 0; r < 8; ++r) {
            int m = mt * 16 + r + 8 * hi;
            out[(size_t)m * DMODEL + n] = acc[j][r];
        }
    }
}

// ---------------------------------------------------------------- launch ----
extern "C" void kernel_launch(void* const* d_in, const int* in_sizes, int n_in,
                              void* d_out, int out_size, void* d_ws, size_t ws_size,
                              hipStream_t stream) {
    const float* x  = (const float*)d_in[0];
    const float* Wq = (const float*)d_in[1];
    const float* Wk = (const float*)d_in[2];
    const float* Wv = (const float*)d_in[3];
    const float* Wo = (const float*)d_in[4];
    float* out = (float*)d_out;

    char* ws = (char*)d_ws;
    const size_t MB = 1u << 20;
    _Float16* xh  = (_Float16*)(ws + 0 * MB);
    _Float16* wqh = (_Float16*)(ws + 8 * MB);
    _Float16* wkh = (_Float16*)(ws + 10 * MB);
    _Float16* wvh = (_Float16*)(ws + 12 * MB);
    _Float16* woh = (_Float16*)(ws + 14 * MB);
    _Float16* qh  = (_Float16*)(ws + 16 * MB);
    _Float16* kh  = (_Float16*)(ws + 24 * MB);
    _Float16* vt  = (_Float16*)(ws + 32 * MB);
    _Float16* oh  = (_Float16*)(ws + 40 * MB);

    int nx4 = (MTOT * DMODEL) / 4;
    k_cvt_f32_f16<<<nx4 / 256, 256, 0, stream>>>(x, xh, nx4);
    int nw4 = (DMODEL * DMODEL) / 4;
    k_cvt_f32_f16<<<nw4 / 256, 256, 0, stream>>>(Wq, wqh, nw4);
    k_cvt_f32_f16<<<nw4 / 256, 256, 0, stream>>>(Wk, wkh, nw4);
    k_cvt_f32_f16<<<nw4 / 256, 256, 0, stream>>>(Wv, wvh, nw4);
    k_cvt_f32_f16<<<nw4 / 256, 256, 0, stream>>>(Wo, woh, nw4);

    k_proj_qkv<<<1536, 256, 0, stream>>>(xh, wqh, wkh, wvh, qh, kh, vt);
    k_attn<<<1024, 128, 0, stream>>>(qh, kh, vt, oh);
    k_proj_out<<<512, 256, 0, stream>>>(oh, woh, out);
}
